// JointAttention_28260884808305
// MI455X (gfx1250) — matrix operand order repaired and checked by reading, not compiled, not run on hardware
//
#include <hip/hip_runtime.h>
#include <hip/hip_bf16.h>

typedef __attribute__((ext_vector_type(16))) _Float16 v16h;
typedef __attribute__((ext_vector_type(8)))  _Float16 v8h;
typedef __attribute__((ext_vector_type(8)))  float    v8f;

#define WMMA_F32_F16(a, b, c) \
  __builtin_amdgcn_wmma_f32_16x16x32_f16(false, (a), false, (b), (short)0, (c), false, false)

__device__ inline v8f v8f_zero() {
  v8f z;
#pragma unroll
  for (int i = 0; i < 8; ++i) z[i] = 0.0f;
  return z;
}

// ---------------------------------------------------------------- constants
#define DIMC   1024
#define NHEAD  16
#define HDIM   64
#define SIMG   2048
#define STXT   256
#define STOT   2304          // SIMG + STXT
#define ROWSI  4096          // B * SIMG
#define ROWST  512           // B * STXT
#define ROWS   4608          // ROWSI + ROWST

// map fused sequence index (b, s in [0,2304)) -> row in the [4608, 1024] buffers
__device__ inline long fused_row(int b, int s) {
  return (s < SIMG) ? ((long)b * SIMG + s) : ((long)ROWSI + (long)b * STXT + (s - SIMG));
}

// ---------------------------------------------------------------- f32 -> f16 cast
__global__ void cast_f16_kernel(const float* __restrict__ src, _Float16* __restrict__ dst, int n) {
  int i = blockIdx.x * blockDim.x + threadIdx.x;
  int stride = gridDim.x * blockDim.x;
  for (; i < n; i += stride) dst[i] = (_Float16)src[i];
}

// ---------------------------------------------------------------- WMMA GEMM:  Y[M,1024] = X[M,1024] @ W[1024,1024]^T + bias
// grid.x = M/32, grid.y = 1024/256, block = 256 (8 waves: 2 along M x 4 along N).
// The 32-row A panel (32x1024 f16 = 64 KB) is staged once into LDS with
// CDNA5 async global->LDS copies (ASYNCcnt), then A fragments come from LDS
// while B streams from L2 pipelined against the WMMAs.
__global__ __launch_bounds__(256) void gemm_bias_kernel(const _Float16* __restrict__ X,
                                                        const _Float16* __restrict__ W,
                                                        const float*    __restrict__ bias,
                                                        float*          __restrict__ Y) {
  __shared__ _Float16 Ax[32][DIMC];    // 64 KB of the 320 KB/WGP LDS

  const int K = DIMC, N = DIMC;
  int t    = threadIdx.x;
  int lane = t & 31;
  int wid  = t >> 5;
  int wm   = wid >> 2;          // 0..1
  int wn   = wid & 3;           // 0..3
  int mblk = blockIdx.x * 32;
  int n0   = blockIdx.y * 256 + wn * 64;
  int half = lane >> 4;         // 0..1
  int ln   = lane & 15;

  // ---- async-stage X[mblk .. mblk+32) x K into LDS: 4096 x b128, 16 per thread
  {
    const _Float16* base = X + (long)mblk * K;
#pragma unroll
    for (int i = 0; i < 16; ++i) {
      int chunk = i * 256 + t;                 // 0..4095, 8 halves each
      int row   = chunk >> 7;                  // 128 chunks per 1024-half row
      int col   = (chunk & 127) << 3;
      unsigned int       ldsoff = (unsigned int)(uintptr_t)&Ax[row][col];   // low 32 bits = LDS offset
      unsigned long long gaddr  = (unsigned long long)(uintptr_t)(base + (long)row * K + col);
      asm volatile("global_load_async_to_lds_b128 %0, %1, off"
                   :: "v"(ldsoff), "v"(gaddr) : "memory");
    }
    asm volatile("s_wait_asynccnt 0" ::: "memory");
  }
  __syncthreads();

  v8f acc[4];
#pragma unroll
  for (int j = 0; j < 4; ++j) acc[j] = v8f_zero();

  const _Float16* arow = &Ax[wm * 16 + ln][0];

  for (int kk = 0; kk < K; kk += 32) {
    __builtin_prefetch(W + (long)(n0 + ln) * K + kk + 512, 0, 1);
    // A fragment from LDS: 16x32, documented striping (half h: K = h*8.. and 16+h*8..)
    v8h lo = *(const v8h*)(arow + kk + half * 8);
    v8h hi = *(const v8h*)(arow + kk + 16 + half * 8);
    v16h a = __builtin_shufflevector(lo, hi, 0, 1, 2, 3, 4, 5, 6, 7,
                                             8, 9, 10, 11, 12, 13, 14, 15);
#pragma unroll
    for (int j = 0; j < 4; ++j) {
      // B fragment: column n = W row n; half h: K = kk+h*8.. and kk+16+h*8..
      int n = n0 + j * 16 + ln;
      const _Float16* wp = W + (long)n * K + kk + half * 8;
      v16h bfrag = __builtin_shufflevector(*(const v8h*)wp, *(const v8h*)(wp + 16), 0, 1, 2, 3, 4, 5, 6, 7, 8, 9, 10, 11, 12, 13, 14, 15);
      acc[j] = WMMA_F32_F16(a, bfrag, acc[j]);
    }
  }

#pragma unroll
  for (int j = 0; j < 4; ++j) {
    int n = n0 + j * 16 + ln;
    float bn = bias[n];
#pragma unroll
    for (int v = 0; v < 8; ++v) {
      int r = mblk + wm * 16 + v + 8 * half;   // C layout: VGPR v holds M = v (lanes 0-15), v+8 (16-31)
      Y[(long)r * N + n] = acc[j][v] + bn;
    }
  }
}

// ---------------------------------------------------------------- per-head RMSNorm + reshape to [B,H,S,64] f16
// block = 256 (8 waves, one (b,h,s) per wave); grid = 2*16*2304/8 = 9216
__global__ __launch_bounds__(256) void norm_qk_kernel(const float* __restrict__ Y,
                                                      const float* __restrict__ gImg,
                                                      const float* __restrict__ gTxt,
                                                      _Float16*    __restrict__ out) {
  int lane = threadIdx.x & 31;
  int w    = threadIdx.x >> 5;
  long idx = (long)blockIdx.x * 8 + w;      // (b*16 + h)*2304 + s
  int  s   = (int)(idx % STOT);
  long bh  = idx / STOT;
  int  b   = (int)(bh / NHEAD);
  int  h   = (int)(bh % NHEAD);
  const float* src = Y + fused_row(b, s) * DIMC + h * HDIM;
  float x0 = src[lane], x1 = src[lane + 32];
  float ss = x0 * x0 + x1 * x1;
#pragma unroll
  for (int m = 16; m >= 1; m >>= 1) ss += __shfl_xor(ss, m, 32);
  float rinv = rsqrtf(ss * (1.0f / 64.0f) + 1e-6f);
  const float* g = (s < SIMG) ? gImg : gTxt;
  _Float16* dst = out + idx * HDIM;
  dst[lane]      = (_Float16)(x0 * rinv * g[lane]);
  dst[lane + 32] = (_Float16)(x1 * rinv * g[lane + 32]);
}

// ---------------------------------------------------------------- V reshape + transpose to [B,H,64,S] f16
__global__ __launch_bounds__(256) void reshape_v_kernel(const float* __restrict__ Yv,
                                                        _Float16*    __restrict__ Vt) {
  int lane = threadIdx.x & 31;
  int w    = threadIdx.x >> 5;
  long idx = (long)blockIdx.x * 8 + w;
  int  s   = (int)(idx % STOT);
  long bh  = idx / STOT;
  int  b   = (int)(bh / NHEAD);
  int  h   = (int)(bh % NHEAD);
  const float* src = Yv + fused_row(b, s) * DIMC + h * HDIM;
  _Float16* dst = Vt + bh * HDIM * STOT;     // [d][s]
  dst[(long)lane * STOT + s]        = (_Float16)src[lane];
  dst[(long)(lane + 32) * STOT + s] = (_Float16)src[lane + 32];
}

// ---------------------------------------------------------------- flash attention, one (b,h,qblock64) per WG
// block = 128 (4 waves, 16 queries each); grid = B*H*(2304/64) = 1152
__global__ __launch_bounds__(128) void attn_kernel(const _Float16* __restrict__ Qh,
                                                   const _Float16* __restrict__ Kh,
                                                   const _Float16* __restrict__ Vt,
                                                   _Float16*       __restrict__ attnH) {
  __shared__ _Float16 pLDS[4][16][32];   // per-wave P staging (C-layout -> A-layout reswizzle)

  int lane = threadIdx.x & 31;
  int w    = threadIdx.x >> 5;
  int half = lane >> 4;
  int ln   = lane & 15;

  int  qb = blockIdx.x % 36;
  long bh = blockIdx.x / 36;
  int  h  = (int)(bh % NHEAD);
  int  b  = (int)(bh / NHEAD);
  int  q0 = qb * 64 + w * 16;

  const _Float16* Q = Qh + bh * (long)STOT * HDIM;
  const _Float16* K = Kh + bh * (long)STOT * HDIM;
  const _Float16* V = Vt + bh * (long)HDIM * STOT;

  // Q A-fragments for d = 0..31 and 32..63 (documented A striping)
  v16h aq[2];
#pragma unroll
  for (int dh = 0; dh < 2; ++dh) {
    const _Float16* p = Q + (long)(q0 + ln) * HDIM + dh * 32 + half * 8;
    v8h lo = *(const v8h*)p;
    v8h hi = *(const v8h*)(p + 16);
    aq[dh] = __builtin_shufflevector(lo, hi, 0, 1, 2, 3, 4, 5, 6, 7,
                                             8, 9, 10, 11, 12, 13, 14, 15);
  }

  v8f O[4];
#pragma unroll
  for (int j = 0; j < 4; ++j) O[j] = v8f_zero();
  float m[8], l[8];
#pragma unroll
  for (int j = 0; j < 8; ++j) { m[j] = -3.0e38f; l[j] = 0.0f; }

  const float scale = 0.125f;   // 1/sqrt(64)

  for (int kb = 0; kb < STOT; kb += 32) {
    // ---- scores S = Q @ K^T : 16 queries x 32 keys, contraction over d=64 (2 wmma per frag)
    v8f sf[2];
#pragma unroll
    for (int f = 0; f < 2; ++f) {
      sf[f] = v8f_zero();
#pragma unroll
      for (int dh = 0; dh < 2; ++dh) {
        int key = kb + f * 16 + ln;
        const _Float16* kp = K + (long)key * HDIM + dh * 32 + half * 8;
        v16h bk = __builtin_shufflevector(*(const v8h*)kp, *(const v8h*)(kp + 16), 0, 1, 2, 3, 4, 5, 6, 7, 8, 9, 10, 11, 12, 13, 14, 15);
        sf[f] = WMMA_F32_F16(aq[dh], bk, sf[f]);
      }
    }

    // ---- online softmax: row M = j + 8*half, its 16 columns live across one 16-lane half
#pragma unroll
    for (int j = 0; j < 8; ++j) {
      float s0 = sf[0][j] * scale;
      float s1 = sf[1][j] * scale;
      float t = fmaxf(s0, s1);
      t = fmaxf(t, __shfl_xor(t, 1, 32));
      t = fmaxf(t, __shfl_xor(t, 2, 32));
      t = fmaxf(t, __shfl_xor(t, 4, 32));
      t = fmaxf(t, __shfl_xor(t, 8, 32));
      float mn = fmaxf(m[j], t);
      float al = __expf(m[j] - mn);
      float p0 = __expf(s0 - mn);
      float p1 = __expf(s1 - mn);
      float r = p0 + p1;
      r += __shfl_xor(r, 1, 32);
      r += __shfl_xor(r, 2, 32);
      r += __shfl_xor(r, 4, 32);
      r += __shfl_xor(r, 8, 32);
      l[j] = l[j] * al + r;
      m[j] = mn;
      O[0][j] *= al; O[1][j] *= al; O[2][j] *= al; O[3][j] *= al;
      int row = j + 8 * half;
      pLDS[w][row][ln]      = (_Float16)p0;
      pLDS[w][row][16 + ln] = (_Float16)p1;
    }
    __syncthreads();

    // ---- O += P @ V : reload P in A-layout, Vt columns are contiguous
    v16h pa;
    {
      const _Float16* p = &pLDS[w][ln][half * 8];
      v8h lo = *(const v8h*)p;
      v8h hi = *(const v8h*)(p + 16);
      pa = __builtin_shufflevector(lo, hi, 0, 1, 2, 3, 4, 5, 6, 7,
                                           8, 9, 10, 11, 12, 13, 14, 15);
    }
#pragma unroll
    for (int nd = 0; nd < 4; ++nd) {
      const _Float16* vp = V + (long)(nd * 16 + ln) * STOT + kb + half * 8;
      v16h bv = __builtin_shufflevector(*(const v8h*)vp, *(const v8h*)(vp + 16), 0, 1, 2, 3, 4, 5, 6, 7, 8, 9, 10, 11, 12, 13, 14, 15);
      O[nd] = WMMA_F32_F16(pa, bv, O[nd]);
    }
    __syncthreads();
  }

  // ---- epilogue: divide by row sums, scatter to fused [4608, 1024] f16 buffer
#pragma unroll
  for (int j = 0; j < 8; ++j) {
    float inv = 1.0f / l[j];
    int q = q0 + j + 8 * half;
    _Float16* dst = attnH + fused_row(b, q) * DIMC + h * HDIM;
#pragma unroll
    for (int nd = 0; nd < 4; ++nd)
      dst[nd * 16 + ln] = (_Float16)(O[nd][j] * inv);
  }
}

// ---------------------------------------------------------------- host side
extern "C" void kernel_launch(void* const* d_in, const int* in_sizes, int n_in,
                              void* d_out, int out_size, void* d_ws, size_t ws_size,
                              hipStream_t stream) {
  (void)in_sizes; (void)n_in; (void)out_size; (void)ws_size;

  char* p = (char*)d_ws;
  auto alloc = [&](size_t bytes) -> void* {
    void* r = (void*)p;
    p += (bytes + 255) & ~(size_t)255;
    return r;
  };

  _Float16* hImg = (_Float16*)alloc((size_t)ROWSI * DIMC * 2);
  _Float16* hTxt = (_Float16*)alloc((size_t)ROWST * DIMC * 2);
  _Float16* wH[8];
  for (int i = 0; i < 8; ++i) wH[i] = (_Float16*)alloc((size_t)DIMC * DIMC * 2);
  float* Yq = (float*)alloc((size_t)ROWS * DIMC * 4);
  float* Yk = (float*)alloc((size_t)ROWS * DIMC * 4);
  float* Yv = (float*)alloc((size_t)ROWS * DIMC * 4);
  _Float16* Qh    = (_Float16*)alloc((size_t)2 * NHEAD * STOT * HDIM * 2);
  _Float16* Kh    = (_Float16*)alloc((size_t)2 * NHEAD * STOT * HDIM * 2);
  _Float16* Vt    = (_Float16*)alloc((size_t)2 * NHEAD * HDIM * STOT * 2);
  _Float16* attnH = (_Float16*)alloc((size_t)ROWS * DIMC * 2);

  // inputs: 0 img_hidden, 1 txt_hidden, 2 wq, 3 bq, 4 wk, 5 bk, 6 wv, 7 bv,
  //         8 waq, 9 baq, 10 wak, 11 bak, 12 wav, 13 bav,
  //         14 wout, 15 bout, 16 waout, 17 baout, 18 gq, 19 gk, 20 gaq, 21 gak
  cast_f16_kernel<<<2048, 256, 0, stream>>>((const float*)d_in[0], hImg, ROWSI * DIMC);
  cast_f16_kernel<<<512, 256, 0, stream>>>((const float*)d_in[1], hTxt, ROWST * DIMC);
  const int widx[8] = {2, 4, 6, 8, 10, 12, 14, 16};
  for (int i = 0; i < 8; ++i)
    cast_f16_kernel<<<1024, 256, 0, stream>>>((const float*)d_in[widx[i]], wH[i], DIMC * DIMC);

  dim3 gI(ROWSI / 32, DIMC / 256);     // 128 x 4
  dim3 gT(ROWST / 32, DIMC / 256);     // 16 x 4

  // QKV projections (img rows 0..4095, txt rows 4096..4607 in each Y buffer)
  gemm_bias_kernel<<<gI, 256, 0, stream>>>(hImg, wH[0], (const float*)d_in[3],  Yq);
  gemm_bias_kernel<<<gT, 256, 0, stream>>>(hTxt, wH[3], (const float*)d_in[9],  Yq + (long)ROWSI * DIMC);
  gemm_bias_kernel<<<gI, 256, 0, stream>>>(hImg, wH[1], (const float*)d_in[5],  Yk);
  gemm_bias_kernel<<<gT, 256, 0, stream>>>(hTxt, wH[4], (const float*)d_in[11], Yk + (long)ROWSI * DIMC);
  gemm_bias_kernel<<<gI, 256, 0, stream>>>(hImg, wH[2], (const float*)d_in[7],  Yv);
  gemm_bias_kernel<<<gT, 256, 0, stream>>>(hTxt, wH[5], (const float*)d_in[13], Yv + (long)ROWSI * DIMC);

  // per-head RMSNorm (Q, K) and V transpose
  norm_qk_kernel<<<9216, 256, 0, stream>>>(Yq, (const float*)d_in[18], (const float*)d_in[20], Qh);
  norm_qk_kernel<<<9216, 256, 0, stream>>>(Yk, (const float*)d_in[19], (const float*)d_in[21], Kh);
  reshape_v_kernel<<<9216, 256, 0, stream>>>(Yv, Vt);

  // fused joint attention
  attn_kernel<<<1152, 128, 0, stream>>>(Qh, Kh, Vt, attnH);

  // output projections straight into d_out (img block then txt block)
  float* out = (float*)d_out;
  gemm_bias_kernel<<<gI, 256, 0, stream>>>(attnH, wH[6], (const float*)d_in[15], out);
  gemm_bias_kernel<<<gT, 256, 0, stream>>>(attnH + (long)ROWSI * DIMC, wH[7],
                                           (const float*)d_in[17], out + (long)ROWSI * DIMC);
}
